// FashionSentenceGenerator_72043781423317
// MI455X (gfx1250) — matrix operand order repaired and checked by reading, not compiled, not run on hardware
//
#include <hip/hip_runtime.h>
#include <hip/hip_bf16.h>

// Problem constants
#define B_   128
#define L_   30
#define M_   10
#define H_   256
#define VN_  32000
#define QW_  ((L_ + 1) * H_)   // 7936
#define NTILE_ (VN_ / 16)      // 2000 vocab tiles of 16 rows

typedef __bf16 bf16_t;
typedef bf16_t v16bf __attribute__((ext_vector_type(16)));
typedef bf16_t v8bf  __attribute__((ext_vector_type(8)));
typedef float  v8f   __attribute__((ext_vector_type(8)));

__device__ __forceinline__ bf16_t to_bf16(float x) {
  union { float f; unsigned u; } v; v.f = x;
  unsigned r = v.u + 0x7FFFu + ((v.u >> 16) & 1u);   // round-to-nearest-even
  union { unsigned short s; bf16_t b; } o; o.s = (unsigned short)(r >> 16);
  return o.b;
}
__device__ __forceinline__ float sigmoidf_(float x) { return 1.0f / (1.0f + expf(-x)); }

__device__ __forceinline__ float block_reduce_sum(float v, float* s_red) {
  int tid = threadIdx.x;
  s_red[tid] = v;
  __syncthreads();
  for (int o = 128; o > 0; o >>= 1) {
    if (tid < o) s_red[tid] += s_red[tid + o];
    __syncthreads();
  }
  float r = s_red[0];
  __syncthreads();
  return r;
}

// ---------------------------------------------------------------------------
// Init: zero outputs (loss + g_history) and per-batch loss accumulators
// ---------------------------------------------------------------------------
__global__ __launch_bounds__(256) void k_init(float* __restrict__ out, int out_n,
                                              float* __restrict__ loss_acc) {
  int i = blockIdx.x * 256 + threadIdx.x;
  if (i < out_n) out[i] = 0.0f;
  if (i < B_) loss_acc[i] = 0.0f;
}

// ---------------------------------------------------------------------------
// Convert W_nv (VN x H fp32) to bf16 (row-major) for WMMA
// ---------------------------------------------------------------------------
__global__ __launch_bounds__(256) void k_convert_wnv(const float* __restrict__ W_nv,
                                                     bf16_t* __restrict__ Wnv_bf) {
  int i = blockIdx.x * 256 + threadIdx.x;
  if (i < VN_ * H_) Wnv_bf[i] = to_bf16(W_nv[i]);
}

// ---------------------------------------------------------------------------
// Setup: masked key/val memories, init_h, hist0, prev_h
// ---------------------------------------------------------------------------
__global__ __launch_bounds__(256) void k_setup(
    const int* __restrict__ keywords, const int* __restrict__ categories,
    const int* __restrict__ memory_size, const float* __restrict__ emb,
    float* __restrict__ key_mem, float* __restrict__ val_mem,
    float* __restrict__ histN, float* __restrict__ histK, float* __restrict__ histV,
    float* __restrict__ h_prev) {
  int b = blockIdx.x, tid = threadIdx.x;
  int ms = memory_size[b];
  __shared__ float s_init[H_];

  size_t mb = (size_t)b * M_ * H_;
  for (int i = tid; i < M_ * H_; i += 256) {
    int m = i / H_, j = i % H_;
    float msk = (m < ms) ? 1.0f : 0.0f;
    key_mem[mb + i] = emb[(size_t)keywords[b * M_ + m] * H_ + j] * msk;
    val_mem[mb + i] = emb[(size_t)categories[b * M_ + m] * H_ + j] * msk;
  }
  __syncthreads();

  float s = 0.0f;
  for (int m = 0; m < M_; ++m)
    s += key_mem[mb + m * H_ + tid] + val_mem[mb + m * H_ + tid];
  float ih = s / (2.0f * (float)ms);
  s_init[tid] = ih;
  h_prev[b * H_ + tid] = ih;
  __syncthreads();

  size_t hb = (size_t)b * L_ * H_;
  for (int i = tid; i < L_ * H_; i += 256) {
    float v = (i < H_) ? s_init[i] : 0.0f;
    histN[hb + i] = v; histK[hb + i] = v; histV[hb + i] = v;
  }
}

// ---------------------------------------------------------------------------
// Per-step small work: one block per batch element.
// History attentions, memory attention, LSTM cell, hN/hK/hV, gate, P_M part.
// ---------------------------------------------------------------------------
__global__ __launch_bounds__(256) void k_small_step(
    int t, const int* __restrict__ sentence, const float* __restrict__ emb,
    const float* __restrict__ W_ct, const float* __restrict__ b_ct,
    const float* __restrict__ W_ih, const float* __restrict__ W_hh,
    const float* __restrict__ b_ih, const float* __restrict__ b_hh,
    const float* __restrict__ W_n, const float* __restrict__ W_k,
    const float* __restrict__ W_v,
    const float* __restrict__ W_g, const float* __restrict__ b_g,
    const float* __restrict__ W_hn, const float* __restrict__ b_hn,
    const float* __restrict__ W_hk, const float* __restrict__ b_hk,
    const float* __restrict__ W_hv, const float* __restrict__ b_hv,
    const float* __restrict__ W_mk, const float* __restrict__ b_mk,
    const float* __restrict__ W_mv, const float* __restrict__ b_mv,
    const float* __restrict__ key_mem, const float* __restrict__ val_mem,
    float* __restrict__ histN, float* __restrict__ histK, float* __restrict__ histV,
    const float* __restrict__ h_prev, float* __restrict__ h_next,
    bf16_t* __restrict__ hN_bf, float* __restrict__ gt_buf,
    float* __restrict__ pmem_buf, float* __restrict__ g_out) {
  __shared__ float s_hist[L_ * H_];
  __shared__ float s_prevh[H_], s_pemb[H_];
  __shared__ float s_cH[3 * H_], s_cMK[H_], s_cMV[H_];
  __shared__ float s_h[H_], s_hK[H_], s_hV[H_];
  __shared__ float s_gates[4 * H_];
  __shared__ float s_red[256];
  __shared__ float s_w[L_], s_wm[M_], s_lk[M_], s_lv[M_];

  int b = blockIdx.x, tid = threadIdx.x;
  s_prevh[tid] = h_prev[b * H_ + tid];
  int ptok = sentence[b * L_ + (t - 1)];
  s_pemb[tid] = emb[(size_t)ptok * H_ + tid];
  __syncthreads();

  // ---- three history attentions ----
  const float* Wh[3] = {W_hn, W_hk, W_hv};
  const float* bh[3] = {b_hn, b_hk, b_hv};
  float* Hs[3] = {histN, histK, histV};
  for (int a = 0; a < 3; ++a) {
    const float* hb = Hs[a] + (size_t)b * L_ * H_;
    for (int i = tid; i < L_ * H_; i += 256) s_hist[i] = hb[i];
    __syncthreads();
    for (int l = 0; l < L_; ++l) {
      const float* wr = Wh[a] + (size_t)l * QW_;
      float part = s_prevh[tid] * wr[tid];
      for (int i = tid; i < L_ * H_; i += 256) part += s_hist[i] * wr[H_ + i];
      float tot = block_reduce_sum(part, s_red);
      if (tid == 0) s_w[l] = tanhf(tot + bh[a][l]);
    }
    __syncthreads();
    float acc = 0.0f;
    for (int l = 0; l < L_; ++l) acc += s_w[l] * s_hist[l * H_ + tid];
    s_cH[a * H_ + tid] = acc;
    __syncthreads();
  }

  // ---- memory attention (q_m = [prev_h, cH]) ----
  size_t mb = (size_t)b * M_ * H_;
  for (int m = 0; m < M_; ++m) {
    const float* wr = W_mk + (size_t)m * 4 * H_;
    float part = s_prevh[tid] * wr[tid] + s_cH[tid] * wr[H_ + tid] +
                 s_cH[H_ + tid] * wr[2 * H_ + tid] + s_cH[2 * H_ + tid] * wr[3 * H_ + tid];
    float tot = block_reduce_sum(part, s_red);
    if (tid == 0) s_wm[m] = tanhf(tot + b_mk[m]);
  }
  __syncthreads();
  {
    float a = 0.0f;
    for (int m = 0; m < M_; ++m) a += s_wm[m] * key_mem[mb + m * H_ + tid];
    s_cMK[tid] = a;
  }
  __syncthreads();
  for (int m = 0; m < M_; ++m) {
    const float* wr = W_mv + (size_t)m * 4 * H_;
    float part = s_prevh[tid] * wr[tid] + s_cH[tid] * wr[H_ + tid] +
                 s_cH[H_ + tid] * wr[2 * H_ + tid] + s_cH[2 * H_ + tid] * wr[3 * H_ + tid];
    float tot = block_reduce_sum(part, s_red);
    if (tid == 0) s_wm[m] = tanhf(tot + b_mv[m]);
  }
  __syncthreads();
  {
    float a = 0.0f;
    for (int m = 0; m < M_; ++m) a += s_wm[m] * val_mem[mb + m * H_ + tid];
    s_cMV[tid] = a;
  }
  __syncthreads();

  // ---- c0 = [cH, cMK, cMV] @ W_ct.T + b_ct ----
  float c0 = b_ct[tid];
  {
    const float* wr = W_ct + (size_t)tid * 5 * H_;
    for (int k = 0; k < 3 * H_; ++k) c0 += s_cH[k] * wr[k];
    for (int k = 0; k < H_; ++k) c0 += s_cMK[k] * wr[3 * H_ + k];
    for (int k = 0; k < H_; ++k) c0 += s_cMV[k] * wr[4 * H_ + k];
  }

  // ---- LSTM gates ----
  for (int g4 = 0; g4 < 4; ++g4) {
    int g = g4 * H_ + tid;
    const float* wi = W_ih + (size_t)g * H_;
    const float* wh = W_hh + (size_t)g * H_;
    float gv = b_ih[g] + b_hh[g];
    for (int k = 0; k < H_; ++k) gv += s_pemb[k] * wi[k] + s_prevh[k] * wh[k];
    s_gates[g] = gv;
  }
  __syncthreads();
  float gi = sigmoidf_(s_gates[tid]);
  float gf = sigmoidf_(s_gates[H_ + tid]);
  float gg = tanhf(s_gates[2 * H_ + tid]);
  float go = sigmoidf_(s_gates[3 * H_ + tid]);
  float c = gf * c0 + gi * gg;
  float h = go * tanhf(c);
  s_h[tid] = h;
  h_next[b * H_ + tid] = h;
  __syncthreads();

  // ---- hN / hK / hV ----
  float hn = 0.0f, hk = 0.0f, hv = 0.0f;
  {
    const float* wn = W_n + (size_t)tid * H_;
    const float* wk = W_k + (size_t)tid * H_;
    const float* wv = W_v + (size_t)tid * H_;
    for (int k = 0; k < H_; ++k) {
      float hh = s_h[k];
      hn += hh * wn[k]; hk += hh * wk[k]; hv += hh * wv[k];
    }
  }
  size_t ho = (size_t)b * L_ * H_ + (size_t)t * H_ + tid;
  histN[ho] = hn; histK[ho] = hk; histV[ho] = hv;
  hN_bf[b * H_ + tid] = to_bf16(hn);
  s_hK[tid] = hk; s_hV[tid] = hv;

  // ---- gate gt ----
  float gtot = block_reduce_sum(s_h[tid] * W_g[tid], s_red);
  float gt = sigmoidf_(gtot + b_g[0]);
  if (tid == 0) { gt_buf[b] = gt; g_out[b * L_ + t] = gt; }
  __syncthreads();

  // ---- memory softmax part of P ----
  if (tid < M_) {
    float lk = 0.0f, lv = 0.0f;
    for (int j = 0; j < H_; ++j) {
      lk += key_mem[mb + tid * H_ + j] * s_hK[j];
      lv += val_mem[mb + tid * H_ + j] * s_hV[j];
    }
    s_lk[tid] = lk; s_lv[tid] = lv;
  }
  __syncthreads();
  if (tid == 0) {
    float mk = -1e30f, mv2 = -1e30f;
    for (int m = 0; m < M_; ++m) { mk = fmaxf(mk, s_lk[m]); mv2 = fmaxf(mv2, s_lv[m]); }
    float sk = 0.0f, sv = 0.0f;
    for (int m = 0; m < M_; ++m) { sk += expf(s_lk[m] - mk); sv += expf(s_lv[m] - mv2); }
    int tok = sentence[b * L_ + t];
    float pm = 0.0f;
    if (tok >= VN_) {
      int mi = tok - VN_;
      if (mi < M_) pm = 0.5f * (expf(s_lk[mi] - mk) / sk + expf(s_lv[mi] - mv2) / sv);
    }
    pmem_buf[b] = pm;
  }
}

// ---------------------------------------------------------------------------
// Vocab projection: WMMA bf16 GEMM with fused bias + tok-logit + partial
// logsumexp.  One block = one 16-row vocab tile, 8 waves cover B=128 columns.
// ---------------------------------------------------------------------------
__global__ __launch_bounds__(256) void k_vocab(
    int t, const bf16_t* __restrict__ Wnv_bf, const bf16_t* __restrict__ hN_bf,
    const float* __restrict__ b_nv, const int* __restrict__ sentence,
    float2* __restrict__ partials, float* __restrict__ tok_logit) {
  __shared__ bf16_t s_A[16 * H_];   // 16 vocab rows x 256 K (8KB)

  int tid = threadIdx.x;
  int m0 = blockIdx.x * 16;

  // cooperative stage of the A tile into LDS (32B per thread)
  {
    int row = tid >> 4, chunk = tid & 15;
    const bf16_t* src = Wnv_bf + (size_t)(m0 + row) * H_ + chunk * 16;
    *(v8bf*)&s_A[row * H_ + chunk * 16] = *(const v8bf*)src;
    *(v8bf*)&s_A[row * H_ + chunk * 16 + 8] = *(const v8bf*)(src + 8);
  }
  __syncthreads();

  int w = tid >> 5;          // wave 0..7 -> batch columns 16w..16w+15
  int lane = tid & 31;
  int row = lane & 15;       // A row / B column within tile
  int half = lane >> 4;

  v8f acc = {};
  const bf16_t* bbase = hN_bf + (size_t)(w * 16 + row) * H_;
#pragma unroll
  for (int k0 = 0; k0 < H_; k0 += 32) {
    // A frag (16x32 bf16): lane<16 holds K k0..k0+15; lane>=16 next 16
    union { v16bf v; v8bf h[2]; } A;
    A.h[0] = *(const v8bf*)&s_A[row * H_ + k0 + half * 16];
    A.h[1] = *(const v8bf*)&s_A[row * H_ + k0 + half * 16 + 8];
    // B frag (32x16 bf16): lane<16 holds K k0..k0+15 of its column; lane>=16 next 16
    v16bf Bf = *(const v16bf*)(bbase + k0 + half * 16);
    acc = __builtin_amdgcn_wmma_f32_16x16x32_bf16(false, A.v, false, Bf,
                                                  (short)0, acc, false, false);
  }

  // epilogue: bias, target-token capture, partial (max, sumexp) per column
  int bb = w * 16 + row;
  int tok = sentence[bb * L_ + t];
  float pmax = -1e30f;
#pragma unroll
  for (int r = 0; r < 8; ++r) {
    int vr = m0 + half * 8 + r;              // C/D layout: lane>=16 holds rows M=8..15
    float val = acc[r] + b_nv[vr];
    acc[r] = val;
    if (vr == tok) tok_logit[bb] = val;      // unique writer across the grid
    pmax = fmaxf(pmax, val);
  }
  float om = __shfl_xor(pmax, 16, 32);
  float tm = fmaxf(pmax, om);
  float se = 0.0f;
#pragma unroll
  for (int r = 0; r < 8; ++r) se += expf(acc[r] - tm);
  se += __shfl_xor(se, 16, 32);
  if (half == 0) partials[(size_t)blockIdx.x * B_ + bb] = make_float2(tm, se);
}

// ---------------------------------------------------------------------------
// Combine per-tile logsumexp partials; accumulate loss per batch element.
// ---------------------------------------------------------------------------
__global__ __launch_bounds__(256) void k_combine(
    int t, const float2* __restrict__ partials, const float* __restrict__ tok_logit,
    const float* __restrict__ gt_buf, const float* __restrict__ pmem_buf,
    const int* __restrict__ sentence, float* __restrict__ loss_acc) {
  __shared__ float sm[256], ss[256];
  int b = blockIdx.x, tid = threadIdx.x;
  float m = -1e30f, s = 0.0f;
  for (int v = tid; v < NTILE_; v += 256) {
    float2 p = partials[(size_t)v * B_ + b];
    float nm = fmaxf(m, p.x);
    s = s * expf(m - nm) + p.y * expf(p.x - nm);
    m = nm;
  }
  sm[tid] = m; ss[tid] = s;
  __syncthreads();
  for (int o = 128; o > 0; o >>= 1) {
    if (tid < o) {
      float m2 = sm[tid + o], s2 = ss[tid + o];
      float nm = fmaxf(sm[tid], m2);
      ss[tid] = ss[tid] * expf(sm[tid] - nm) + s2 * expf(m2 - nm);
      sm[tid] = nm;
    }
    __syncthreads();
  }
  if (tid == 0) {
    float lse = sm[0] + logf(ss[0]);
    int tok = sentence[b * L_ + t];
    float gt = gt_buf[b];
    float p = (tok < VN_) ? gt * expf(tok_logit[b] - lse)
                          : (1.0f - gt) * pmem_buf[b];
    loss_acc[b] += -logf(p);
  }
}

// ---------------------------------------------------------------------------
// Final: sum per-batch losses into d_out[0] (deterministic)
// ---------------------------------------------------------------------------
__global__ __launch_bounds__(128) void k_final(const float* __restrict__ loss_acc,
                                               float* __restrict__ out) {
  __shared__ float s_red[128];
  int tid = threadIdx.x;
  s_red[tid] = loss_acc[tid];
  __syncthreads();
  for (int o = 64; o > 0; o >>= 1) {
    if (tid < o) s_red[tid] += s_red[tid + o];
    __syncthreads();
  }
  if (tid == 0) out[0] = s_red[0];
}

// ---------------------------------------------------------------------------
extern "C" void kernel_launch(void* const* d_in, const int* in_sizes, int n_in,
                              void* d_out, int out_size, void* d_ws, size_t ws_size,
                              hipStream_t stream) {
  const int* sentence = (const int*)d_in[0];
  const int* keywords = (const int*)d_in[1];
  const int* categories = (const int*)d_in[2];
  const int* memory_size = (const int*)d_in[3];
  // d_in[4] = use_teacher_forcing (always 1)
  const float* emb  = (const float*)d_in[5];
  const float* W_ct = (const float*)d_in[6];
  const float* b_ct = (const float*)d_in[7];
  const float* W_ih = (const float*)d_in[8];
  const float* W_hh = (const float*)d_in[9];
  const float* b_ih = (const float*)d_in[10];
  const float* b_hh = (const float*)d_in[11];
  const float* W_n  = (const float*)d_in[12];
  const float* W_k  = (const float*)d_in[13];
  const float* W_v  = (const float*)d_in[14];
  const float* W_nv = (const float*)d_in[15];
  const float* b_nv = (const float*)d_in[16];
  const float* W_g  = (const float*)d_in[17];
  const float* b_g  = (const float*)d_in[18];
  const float* W_hn = (const float*)d_in[19];
  const float* b_hn = (const float*)d_in[20];
  const float* W_hk = (const float*)d_in[21];
  const float* b_hk = (const float*)d_in[22];
  const float* W_hv = (const float*)d_in[23];
  const float* b_hv = (const float*)d_in[24];
  const float* W_mk = (const float*)d_in[25];
  const float* b_mk = (const float*)d_in[26];
  const float* W_mv = (const float*)d_in[27];
  const float* b_mv = (const float*)d_in[28];
  float* out = (float*)d_out;

  // workspace carve (256B aligned)
  size_t off = 0;
  auto carve = [&](size_t bytes) -> void* {
    void* p = (char*)d_ws + off;
    off += (bytes + 255) & ~(size_t)255;
    return p;
  };
  float* key_mem = (float*)carve((size_t)B_ * M_ * H_ * 4);
  float* val_mem = (float*)carve((size_t)B_ * M_ * H_ * 4);
  float* histN   = (float*)carve((size_t)B_ * L_ * H_ * 4);
  float* histK   = (float*)carve((size_t)B_ * L_ * H_ * 4);
  float* histV   = (float*)carve((size_t)B_ * L_ * H_ * 4);
  float* h0      = (float*)carve((size_t)B_ * H_ * 4);
  float* h1      = (float*)carve((size_t)B_ * H_ * 4);
  bf16_t* hN_bf  = (bf16_t*)carve((size_t)B_ * H_ * 2);
  bf16_t* Wnv_bf = (bf16_t*)carve((size_t)VN_ * H_ * 2);
  float2* partials = (float2*)carve((size_t)NTILE_ * B_ * 8);
  float* tok_logit = (float*)carve(B_ * 4);
  float* gt_buf    = (float*)carve(B_ * 4);
  float* pmem_buf  = (float*)carve(B_ * 4);
  float* loss_acc  = (float*)carve(B_ * 4);
  (void)n_in; (void)in_sizes; (void)ws_size;

  k_init<<<16, 256, 0, stream>>>(out, out_size, loss_acc);
  k_convert_wnv<<<(VN_ * H_) / 256, 256, 0, stream>>>(W_nv, Wnv_bf);
  k_setup<<<B_, 256, 0, stream>>>(keywords, categories, memory_size, emb,
                                  key_mem, val_mem, histN, histK, histV, h0);

  float* hprev = h0;
  float* hnext = h1;
  for (int t = 1; t < L_; ++t) {
    k_small_step<<<B_, 256, 0, stream>>>(
        t, sentence, emb, W_ct, b_ct, W_ih, W_hh, b_ih, b_hh, W_n, W_k, W_v,
        W_g, b_g, W_hn, b_hn, W_hk, b_hk, W_hv, b_hv, W_mk, b_mk, W_mv, b_mv,
        key_mem, val_mem, histN, histK, histV, hprev, hnext, hN_bf,
        gt_buf, pmem_buf, out + 1);
    k_vocab<<<NTILE_, 256, 0, stream>>>(t, Wnv_bf, hN_bf, b_nv, sentence,
                                        partials, tok_logit);
    k_combine<<<B_, 256, 0, stream>>>(t, partials, tok_logit, gt_buf, pmem_buf,
                                      sentence, loss_acc);
    float* tmp = hprev; hprev = hnext; hnext = tmp;
  }
  k_final<<<1, 128, 0, stream>>>(loss_acc, out);
}
